// GaussianInteractionBlock_20048907338323
// MI455X (gfx1250) — matrix operand order repaired and checked by reading, not compiled, not run on hardware
//
#include <hip/hip_runtime.h>
#include <math.h>

// ---------------------------------------------------------------------------
// MI455X (gfx1250): all GEMM-shaped math on v_wmma_f32_16x16x32_bf16.
// Operands pre-converted to bf16 (weights pre-transposed to [N][K]) so WMMA
// fragments are contiguous 16B chunks -> direct global_load_b128.
// Flash attention stages K/V tiles into LDS with global_load_async_to_lds_b128
// (double-buffered, ASYNCcnt) since all 8 waves of a block share them.
// ---------------------------------------------------------------------------

#define C_DIM 256
#define HEADS 8
#define HDIM  32
#define NTOK  4096

typedef __attribute__((ext_vector_type(16))) __bf16 v16bf;
typedef __attribute__((ext_vector_type(2)))  __bf16 v2bf;
typedef __attribute__((ext_vector_type(8)))  float  v8f;

union BFrag { v16bf v; unsigned int u[8]; };

__device__ __forceinline__ unsigned short f2bf(float f) {
  unsigned int u = __builtin_bit_cast(unsigned int, f);
  u += 0x7FFFu + ((u >> 16) & 1u);            // RNE
  return (unsigned short)(u >> 16);
}
__device__ __forceinline__ unsigned int pack2(float a, float b) {
#if __has_builtin(__builtin_amdgcn_cvt_pk_bf16_f32)
  v2bf r = __builtin_amdgcn_cvt_pk_bf16_f32(a, b);   // v_cvt_pk_bf16_f32
  return __builtin_bit_cast(unsigned int, r);
#else
  return (unsigned int)f2bf(a) | ((unsigned int)f2bf(b) << 16);
#endif
}
__device__ __forceinline__ v8f wmma_bf16(const BFrag& a, const BFrag& b, v8f c) {
  return __builtin_amdgcn_wmma_f32_16x16x32_bf16(false, a.v, false, b.v,
                                                 (short)0, c, false, false);
}
__device__ __forceinline__ float gelu_exact(float x) {
  return 0.5f * x * (1.0f + erff(x * 0.7071067811865476f));
}
__device__ __forceinline__ void wait_async0() {
#if __has_builtin(__builtin_amdgcn_s_wait_asynccnt)
  __builtin_amdgcn_s_wait_asynccnt(0);
#else
  asm volatile("s_wait_asynccnt 0x0" ::: "memory");
#endif
}
// A-fragment (16x32) per-lane K pair start (contiguous: k = lh*16 + 2v)
__device__ __forceinline__ int a_kpair(int v, int lh) {
  return lh * 16 + 2 * v;
}
// A fragment from bf16 row-major: two contiguous 16B chunks per lane.
__device__ __forceinline__ void load_af(BFrag& f, const unsigned short* rowk0, int lh) {
  const uint4* p = (const uint4*)rowk0;  // &A[row*K + k0], 16B aligned
  uint4 a = p[2 * lh];        // k = lh*16 .. +8
  uint4 b = p[2 * lh + 1];    // k = lh*16 + 8 .. +8
  f.u[0] = a.x; f.u[1] = a.y; f.u[2] = a.z; f.u[3] = a.w;
  f.u[4] = b.x; f.u[5] = b.y; f.u[6] = b.z; f.u[7] = b.w;
}
// B fragment from K-major operand: 16 contiguous bf16 per lane.
__device__ __forceinline__ void load_bf(BFrag& f, const unsigned short* pk) {
  const uint4* p = (const uint4*)pk;     // 16B aligned
  uint4 a = p[0], b = p[1];
  f.u[0] = a.x; f.u[1] = a.y; f.u[2] = a.z; f.u[3] = a.w;
  f.u[4] = b.x; f.u[5] = b.y; f.u[6] = b.z; f.u[7] = b.w;
}
// One lane's 16B async copy global -> LDS (ASYNCcnt tracked).
__device__ __forceinline__ void async_cp16(const unsigned short* g, unsigned short* l) {
  unsigned lds = (unsigned)(size_t)l;        // low 32 bits = LDS offset
  asm volatile("global_load_async_to_lds_b128 %0, %1, off"
               :: "v"(lds), "v"(g) : "memory");
}

// ---------------------------------------------------------------------------
// Conversions (one-time per tensor)
// ---------------------------------------------------------------------------
__global__ __launch_bounds__(256)
void cvt_bf16_k(const float* __restrict__ in, unsigned short* __restrict__ out) {
  int i = blockIdx.x * 256 + threadIdx.x;            // over total/2
  ((unsigned int*)out)[i] = pack2(in[2 * i], in[2 * i + 1]);
}
// W [K][N] fp32 -> Wt [N][K] bf16 (K-major for B fragments)
__global__ __launch_bounds__(256)
void cvt_wT_k(const float* __restrict__ in, unsigned short* __restrict__ out,
              int K2, int N) {
  int i = blockIdx.x * 256 + threadIdx.x;            // over N*K/2
  int n = i / K2, kp = i - n * K2;
  ((unsigned int*)out)[(size_t)n * K2 + kp] =
      pack2(in[(size_t)(2 * kp) * N + n], in[(size_t)(2 * kp + 1) * N + n]);
}

// ---------------------------------------------------------------------------
// GEMM: Out[M,N] = act(A @ Wt^T + bias) (+res). A:[M][K] bf16, Wt:[N][K] bf16.
// Block = 8 waves; wave computes a 32x32 tile (4 WMMA / K-step); block 64x128.
// OMODE: 0 = fp32 (+optional residual), 1 = bf16 row-major, 2 = bf16 [N][M].
// ---------------------------------------------------------------------------
template <int ACT, int OMODE>
__global__ __launch_bounds__(256)
void gemm_bf16(const unsigned short* __restrict__ A,
               const unsigned short* __restrict__ Wt,
               const float* __restrict__ bias, const float* __restrict__ res,
               void* __restrict__ OutP, int M, int N, int K) {
  const int tid = threadIdx.x;
  const int wv = tid >> 5, lane = tid & 31;
  const int lh = lane >> 4, l16 = lane & 15;
  const int wm = wv & 1, wn = wv >> 1;
  const int m0 = blockIdx.y * 64 + wm * 32;
  const int n0 = blockIdx.x * 128 + wn * 32;

  v8f c00 = {0.f,0.f,0.f,0.f,0.f,0.f,0.f,0.f};
  v8f c01 = c00, c10 = c00, c11 = c00;
  const unsigned short* a0p = A + (size_t)(m0 + l16) * K;
  const unsigned short* a1p = A + (size_t)(m0 + 16 + l16) * K;
  const unsigned short* b0p = Wt + (size_t)(n0 + l16) * K + lh * 16;
  const unsigned short* b1p = Wt + (size_t)(n0 + 16 + l16) * K + lh * 16;

  for (int k0 = 0; k0 < K; k0 += 32) {
    BFrag a0, a1, b0, b1;
    load_af(a0, a0p + k0, lh);
    load_af(a1, a1p + k0, lh);
    load_bf(b0, b0p + k0);
    load_bf(b1, b1p + k0);
    if (k0 + 32 < K) {                                  // global_prefetch_b8
      __builtin_prefetch(a0p + k0 + 32, 0, 1);
      __builtin_prefetch(b0p + k0 + 32, 0, 1);
    }
    c00 = wmma_bf16(a0, b0, c00);
    c01 = wmma_bf16(a0, b1, c01);
    c10 = wmma_bf16(a1, b0, c10);
    c11 = wmma_bf16(a1, b1, c11);
  }
#pragma unroll
  for (int t = 0; t < 4; ++t) {
    v8f c = (t == 0) ? c00 : (t == 1) ? c01 : (t == 2) ? c10 : c11;
    int mb = m0 + (t >> 1) * 16;
    int nb = n0 + (t & 1) * 16;
#pragma unroll
    for (int r = 0; r < 8; ++r) {
      int row = mb + r + 8 * lh;
      int col = nb + l16;
      float x = c[r] + bias[col];
      if (ACT == 1) x = gelu_exact(x);
      if (OMODE == 0) {
        if (res) x += res[(size_t)row * N + col];
        ((float*)OutP)[(size_t)row * N + col] = x;
      } else if (OMODE == 1) {
        ((unsigned short*)OutP)[(size_t)row * N + col] = f2bf(x);
      } else {
        ((unsigned short*)OutP)[(size_t)col * M + row] = f2bf(x);
      }
    }
  }
}

// ---------------------------------------------------------------------------
// LayerNorm over C=256 -> bf16 output (always feeds a WMMA GEMM).
// ---------------------------------------------------------------------------
__global__ __launch_bounds__(256)
void layernorm_k(const float* __restrict__ in, const float* __restrict__ g,
                 const float* __restrict__ b, unsigned short* __restrict__ out) {
  int wv = threadIdx.x >> 5, lane = threadIdx.x & 31;
  int row = blockIdx.x * 8 + wv;
  const float* p = in + (size_t)row * C_DIM;
  float x[8]; float s = 0.f;
#pragma unroll
  for (int j = 0; j < 8; ++j) { x[j] = p[lane + 32 * j]; s += x[j]; }
#pragma unroll
  for (int o = 16; o > 0; o >>= 1) s += __shfl_xor(s, o, 32);
  float mean = s * (1.0f / 256.0f);
  float vv = 0.f;
#pragma unroll
  for (int j = 0; j < 8; ++j) { float d = x[j] - mean; vv += d * d; }
#pragma unroll
  for (int o = 16; o > 0; o >>= 1) vv += __shfl_xor(vv, o, 32);
  float inv = rsqrtf(vv * (1.0f / 256.0f) + 1e-5f);
#pragma unroll
  for (int j = 0; j < 8; ++j) {
    int c = lane + 32 * j;
    out[(size_t)row * C_DIM + c] = f2bf((x[j] - mean) * inv * g[c] + b[c]);
  }
}

// ---------------------------------------------------------------------------
// Cross attention, flash-style (no scale per reference). Q,K: [4096][256] bf16,
// V transposed [256][4096] bf16. Wave = 16 query rows for one head; the block's
// 8 waves share K/V tiles -> async-staged into LDS, double-buffered, 64 keys
// per iteration (8 WMMA / iter). grid = (NTOK/128, HEADS), block = 256.
// ---------------------------------------------------------------------------
__global__ __launch_bounds__(256)
void cross_attn_flash(const unsigned short* __restrict__ Q,
                      const unsigned short* __restrict__ Kx,
                      const unsigned short* __restrict__ Vt,
                      unsigned short* __restrict__ O) {
  __shared__ __align__(16) unsigned short Kl[2][64][32];  // [key][d]
  __shared__ __align__(16) unsigned short Vl[2][32][64];  // [d][key]
  __shared__ __align__(4)  unsigned short Pl[8][16][66];
  const int tid = threadIdx.x;
  const int wv = tid >> 5, lane = tid & 31;
  const int lh = lane >> 4, l16 = lane & 15;
  const int h = blockIdx.y;
  const int r0 = blockIdx.x * 128 + wv * 16;

  // per-thread async-stage mapping (16B chunks)
  const int krow = tid >> 2, kchunk = (tid & 3) * 8;   // 64 rows x 4 chunks
  const int vrow = tid >> 3, vchunk = (tid & 7) * 8;   // 32 rows x 8 chunks

  BFrag qf;
  load_af(qf, Q + (size_t)(r0 + l16) * C_DIM + h * HDIM, lh);

  float m[8], l[8];
  v8f o0 = {0.f,0.f,0.f,0.f,0.f,0.f,0.f,0.f};
  v8f o1 = o0;
#pragma unroll
  for (int r = 0; r < 8; ++r) { m[r] = -1e30f; l[r] = 0.f; }

  // stage tile 0
  async_cp16(Kx + (size_t)(0 + krow) * C_DIM + h * HDIM + kchunk, &Kl[0][krow][kchunk]);
  async_cp16(Vt + (size_t)(h * HDIM + vrow) * NTOK + 0 + vchunk,  &Vl[0][vrow][vchunk]);

  for (int it = 0; it < NTOK / 64; ++it) {
    const int buf = it & 1;
    wait_async0();
    __syncthreads();                       // tile ready; prior buf consumed
    if (it + 1 < NTOK / 64) {
      int j1 = (it + 1) * 64;
      async_cp16(Kx + (size_t)(j1 + krow) * C_DIM + h * HDIM + kchunk,
                 &Kl[buf ^ 1][krow][kchunk]);
      async_cp16(Vt + (size_t)(h * HDIM + vrow) * NTOK + j1 + vchunk,
                 &Vl[buf ^ 1][vrow][vchunk]);
    }
    // S = Q @ K^T  (4 tiles of 16 keys)
    v8f z = {0.f,0.f,0.f,0.f,0.f,0.f,0.f,0.f};
    v8f s[4];
#pragma unroll
    for (int kt = 0; kt < 4; ++kt) {
      BFrag bk;
      load_bf(bk, &Kl[buf][kt * 16 + l16][lh * 16]);   // ds_load_b128
      s[kt] = wmma_bf16(qf, bk, z);
    }
    // online softmax over 64 keys
#pragma unroll
    for (int r = 0; r < 8; ++r) {
      float mx = fmaxf(fmaxf(s[0][r], s[1][r]), fmaxf(s[2][r], s[3][r]));
#pragma unroll
      for (int o = 8; o > 0; o >>= 1) mx = fmaxf(mx, __shfl_xor(mx, o, 32));
      float mn = fmaxf(m[r], mx);
      float alpha = __expf(m[r] - mn);
      float rs = 0.f;
      int prow = r + 8 * lh;
#pragma unroll
      for (int kt = 0; kt < 4; ++kt) {
        float pv = __expf(s[kt][r] - mn);
        rs += pv;
        Pl[wv][prow][kt * 16 + l16] = f2bf(pv);
      }
#pragma unroll
      for (int o = 8; o > 0; o >>= 1) rs += __shfl_xor(rs, o, 32);
      l[r] = l[r] * alpha + rs;
      m[r] = mn;
      o0[r] *= alpha; o1[r] *= alpha;
    }
    asm volatile("s_wait_dscnt 0x0" ::: "memory");  // LDS RAW fence (same wave)
    BFrag p0, p1;
#pragma unroll
    for (int v = 0; v < 8; ++v) {
      int ka = a_kpair(v, lh);
      p0.u[v] = *(const unsigned int*)&Pl[wv][l16][ka];
      p1.u[v] = *(const unsigned int*)&Pl[wv][l16][32 + ka];
    }
#pragma unroll
    for (int kb = 0; kb < 2; ++kb) {
      BFrag vf0, vf1;  // V^T: contraction = key, n = d
      load_bf(vf0, &Vl[buf][l16][kb * 32 + lh * 16]);
      load_bf(vf1, &Vl[buf][16 + l16][kb * 32 + lh * 16]);
      o0 = wmma_bf16(kb ? p1 : p0, vf0, o0);
      o1 = wmma_bf16(kb ? p1 : p0, vf1, o1);
    }
  }
#pragma unroll
  for (int r = 0; r < 8; ++r) {
    int row = r0 + r + 8 * lh;
    float inv = 1.0f / l[r];
    O[(size_t)row * C_DIM + h * HDIM + l16]      = f2bf(o0[r] * inv);
    O[(size_t)row * C_DIM + h * HDIM + 16 + l16] = f2bf(o1[r] * inv);
  }
}

// ---------------------------------------------------------------------------
// Swin window attention. QK: [4096][512] bf16 (q cols h*32, k cols 256+h*32);
// VtW: [256][4096] bf16 (v transposed, row = h*32+d, col = token).
// grid = 64 windows, block = 8 waves (wave = head). Scale folded into logits.
// ---------------------------------------------------------------------------
__global__ __launch_bounds__(256)
void window_attn(const unsigned short* __restrict__ QK,
                 const unsigned short* __restrict__ VtW,
                 const float* __restrict__ rpb, unsigned short* __restrict__ O) {
  __shared__ __align__(4) unsigned short Pl[8][16][66];
  int wv = threadIdx.x >> 5, lane = threadIdx.x & 31;
  int lh = lane >> 4, l16 = lane & 15;
  int h = wv;
  int w = blockIdx.x;
  int wy = w >> 3, wx = w & 7;
  const float scale = 0.17677669529663687f;   // 32^-0.5
  auto grow = [&](int t) {
    int iy = t >> 3, ix = t & 7;
    return (wy * 8 + iy) * 64 + wx * 8 + ix;
  };
  v8f z = {0.f,0.f,0.f,0.f,0.f,0.f,0.f,0.f};

  for (int qt = 0; qt < 4; ++qt) {
    BFrag qf;
    load_af(qf, QK + (size_t)grow(qt * 16 + l16) * 512 + h * HDIM, lh);
    v8f s[4];
#pragma unroll
    for (int kt = 0; kt < 4; ++kt) {
      BFrag bk;
      load_bf(bk, QK + (size_t)grow(kt * 16 + l16) * 512 + 256 + h * HDIM + lh * 16);
      s[kt] = wmma_bf16(qf, bk, z);
    }
#pragma unroll
    for (int r = 0; r < 8; ++r) {
      int qn = qt * 16 + r + 8 * lh;
      float pv[4];
      float mx = -1e30f;
#pragma unroll
      for (int kt = 0; kt < 4; ++kt) {
        int kn = kt * 16 + l16;
        int rpi = ((qn >> 3) - (kn >> 3) + 7) * 15 + ((qn & 7) - (kn & 7) + 7);
        pv[kt] = s[kt][r] * scale + rpb[rpi * HEADS + h];
        mx = fmaxf(mx, pv[kt]);
      }
#pragma unroll
      for (int o = 8; o > 0; o >>= 1) mx = fmaxf(mx, __shfl_xor(mx, o, 32));
      float sum = 0.f;
#pragma unroll
      for (int kt = 0; kt < 4; ++kt) { pv[kt] = __expf(pv[kt] - mx); sum += pv[kt]; }
#pragma unroll
      for (int o = 8; o > 0; o >>= 1) sum += __shfl_xor(sum, o, 32);
      float inv = 1.0f / sum;
      int prow = r + 8 * lh;
#pragma unroll
      for (int kt = 0; kt < 4; ++kt)
        Pl[wv][prow][kt * 16 + l16] = f2bf(pv[kt] * inv);
    }
    asm volatile("s_wait_dscnt 0x0" ::: "memory");
    BFrag p0, p1;
#pragma unroll
    for (int v = 0; v < 8; ++v) {
      int ka = a_kpair(v, lh);
      p0.u[v] = *(const unsigned int*)&Pl[wv][l16][ka];
      p1.u[v] = *(const unsigned int*)&Pl[wv][l16][32 + ka];
    }
    v8f o0 = z, o1 = z;
#pragma unroll
    for (int kb = 0; kb < 2; ++kb) {
      BFrag vf0, vf1;
#pragma unroll
      for (int v = 0; v < 8; ++v) {
        int kk = kb * 32 + lh * 16 + 2 * v;
        int tok = grow(kk);    // kk even -> (tok, tok+1) contiguous
        vf0.u[v] = *(const unsigned int*)(VtW + (size_t)(h * HDIM + l16) * NTOK + tok);
        vf1.u[v] = *(const unsigned int*)(VtW + (size_t)(h * HDIM + 16 + l16) * NTOK + tok);
      }
      o0 = wmma_bf16(kb ? p1 : p0, vf0, o0);
      o1 = wmma_bf16(kb ? p1 : p0, vf1, o1);
    }
#pragma unroll
    for (int r = 0; r < 8; ++r) {
      int row = grow(qt * 16 + r + 8 * lh);
      O[(size_t)row * C_DIM + h * HDIM + l16]      = f2bf(o0[r]);
      O[(size_t)row * C_DIM + h * HDIM + 16 + l16] = f2bf(o1[r]);
    }
  }
}

// ---------------------------------------------------------------------------
// x1[n][c] = in[(n-4) mod 4096][(c-4) mod 256] + in[n][c]
// ---------------------------------------------------------------------------
__global__ __launch_bounds__(256)
void roll_add_k(const float* __restrict__ in, float* __restrict__ out) {
  int i = blockIdx.x * 256 + threadIdx.x;
  int n = i >> 8, c = i & 255;
  int ns = (n + NTOK - 4) & (NTOK - 1);
  int cs = (c + 256 - 4) & 255;
  out[i] = in[(size_t)ns * 256 + cs] + in[i];
}

// ---------------------------------------------------------------------------
extern "C" void kernel_launch(void* const* d_in, const int* in_sizes, int n_in,
                              void* d_out, int out_size, void* d_ws, size_t ws_size,
                              hipStream_t stream) {
  (void)in_sizes; (void)n_in; (void)out_size; (void)ws_size;
  const float* x      = (const float*)d_in[0];
  const float* mlpo   = (const float*)d_in[1];
  const float* cq_w   = (const float*)d_in[2];
  const float* cq_b   = (const float*)d_in[3];
  const float* ck_w   = (const float*)d_in[4];
  const float* ck_b   = (const float*)d_in[5];
  const float* cv_w   = (const float*)d_in[6];
  const float* cv_b   = (const float*)d_in[7];
  const float* cp_w   = (const float*)d_in[8];
  const float* cp_b   = (const float*)d_in[9];
  const float* s1_n1g = (const float*)d_in[10];
  const float* s1_n1b = (const float*)d_in[11];
  const float* s1_qkvw= (const float*)d_in[12];
  const float* s1_qkvb= (const float*)d_in[13];
  const float* s1_rpb = (const float*)d_in[14];
  const float* s1_pw  = (const float*)d_in[15];
  const float* s1_pb  = (const float*)d_in[16];
  const float* s1_n2g = (const float*)d_in[17];
  const float* s1_n2b = (const float*)d_in[18];
  const float* s1_f1w = (const float*)d_in[19];
  const float* s1_f1b = (const float*)d_in[20];
  const float* s1_f2w = (const float*)d_in[21];
  const float* s1_f2b = (const float*)d_in[22];
  const float* s2_n2g = (const float*)d_in[30];
  const float* s2_n2b = (const float*)d_in[31];
  const float* s2_f1w = (const float*)d_in[32];
  const float* s2_f1b = (const float*)d_in[33];
  const float* s2_f2w = (const float*)d_in[34];
  const float* s2_f2b = (const float*)d_in[35];

  // ---- workspace carving (all 256B aligned) ----
  char* p = (char*)d_ws;
  auto alloc = [&](size_t bytes) {
    char* r = p; p += (bytes + 255) & ~(size_t)255; return r;
  };
  const size_t U = (size_t)NTOK * C_DIM;                // 1M elements
  float* t0 = (float*)alloc(U * 4);
  float* t1 = (float*)alloc(U * 4);
  float* t2 = (float*)alloc(U * 4);
  unsigned short* xb   = (unsigned short*)alloc(U * 2);
  unsigned short* mb   = (unsigned short*)alloc(U * 2);
  unsigned short* qb   = (unsigned short*)alloc(U * 2);
  unsigned short* kb   = (unsigned short*)alloc(U * 2);
  unsigned short* vt   = (unsigned short*)alloc(U * 2); // [256][4096]
  unsigned short* ob   = (unsigned short*)alloc(U * 2);
  unsigned short* lnb  = (unsigned short*)alloc(U * 2);
  unsigned short* qkb  = (unsigned short*)alloc(U * 2 * 2);  // [4096][512]
  unsigned short* vtw  = (unsigned short*)alloc(U * 2);      // [256][4096]
  unsigned short* h1b  = (unsigned short*)alloc(U * 2 * 4);  // [4096][1024]
  unsigned short* cqT  = (unsigned short*)alloc(256 * 256 * 2);
  unsigned short* ckT  = (unsigned short*)alloc(256 * 256 * 2);
  unsigned short* cvT  = (unsigned short*)alloc(256 * 256 * 2);
  unsigned short* cpT  = (unsigned short*)alloc(256 * 256 * 2);
  unsigned short* qkvT = (unsigned short*)alloc(768 * 256 * 2);
  unsigned short* pwT  = (unsigned short*)alloc(256 * 256 * 2);
  unsigned short* f1T  = (unsigned short*)alloc(1024 * 256 * 2);
  unsigned short* f2T  = (unsigned short*)alloc(256 * 1024 * 2);
  unsigned short* g1T  = (unsigned short*)alloc(1024 * 256 * 2);
  unsigned short* g2T  = (unsigned short*)alloc(256 * 1024 * 2);

  dim3 blk(256);
  dim3 gP(C_DIM / 128, NTOK / 64);     // N=256
  dim3 gQK(512 / 128, NTOK / 64);      // N=512
  dim3 gF1(1024 / 128, NTOK / 64);     // N=1024
  const int CB = (NTOK * C_DIM / 2) / 256;  // act conversion blocks

  // ---- one-time conversions ----
  cvt_bf16_k<<<CB, blk, 0, stream>>>(x, xb);
  cvt_bf16_k<<<CB, blk, 0, stream>>>(mlpo, mb);
  cvt_wT_k<<<(256 * 128) / 256, blk, 0, stream>>>(cq_w, cqT, 128, 256);
  cvt_wT_k<<<(256 * 128) / 256, blk, 0, stream>>>(ck_w, ckT, 128, 256);
  cvt_wT_k<<<(256 * 128) / 256, blk, 0, stream>>>(cv_w, cvT, 128, 256);
  cvt_wT_k<<<(256 * 128) / 256, blk, 0, stream>>>(cp_w, cpT, 128, 256);
  cvt_wT_k<<<(768 * 128) / 256, blk, 0, stream>>>(s1_qkvw, qkvT, 128, 768);
  cvt_wT_k<<<(256 * 128) / 256, blk, 0, stream>>>(s1_pw, pwT, 128, 256);
  cvt_wT_k<<<(1024 * 128) / 256, blk, 0, stream>>>(s1_f1w, f1T, 128, 1024);
  cvt_wT_k<<<(256 * 512) / 256, blk, 0, stream>>>(s1_f2w, f2T, 512, 256);
  cvt_wT_k<<<(1024 * 128) / 256, blk, 0, stream>>>(s2_f1w, g1T, 128, 1024);
  cvt_wT_k<<<(256 * 512) / 256, blk, 0, stream>>>(s2_f2w, g2T, 512, 256);

  // ---- cross attention ----
  gemm_bf16<0,1><<<gP, blk, 0, stream>>>(xb, cqT, cq_b, nullptr, qb, NTOK, 256, 256);
  gemm_bf16<0,1><<<gP, blk, 0, stream>>>(mb, ckT, ck_b, nullptr, kb, NTOK, 256, 256);
  gemm_bf16<0,2><<<gP, blk, 0, stream>>>(mb, cvT, cv_b, nullptr, vt, NTOK, 256, 256);
  cross_attn_flash<<<dim3(NTOK / 128, HEADS), blk, 0, stream>>>(qb, kb, vt, ob);
  gemm_bf16<0,0><<<gP, blk, 0, stream>>>(ob, cpT, cp_b, x, t1, NTOK, 256, 256); // y0

  // ---- swin block 1 (no shift) ----
  layernorm_k<<<NTOK / 8, blk, 0, stream>>>(t1, s1_n1g, s1_n1b, lnb);
  gemm_bf16<0,1><<<gQK, blk, 0, stream>>>(lnb, qkvT, s1_qkvb, nullptr, qkb, NTOK, 512, 256);
  gemm_bf16<0,2><<<gP, blk, 0, stream>>>(lnb, qkvT + 512 * 256, s1_qkvb + 512,
                                         nullptr, vtw, NTOK, 256, 256);
  window_attn<<<64, blk, 0, stream>>>(qkb, vtw, s1_rpb, ob);
  gemm_bf16<0,0><<<gP, blk, 0, stream>>>(ob, pwT, s1_pb, t1, t2, NTOK, 256, 256); // y1
  layernorm_k<<<NTOK / 8, blk, 0, stream>>>(t2, s1_n2g, s1_n2b, lnb);
  gemm_bf16<1,1><<<gF1, blk, 0, stream>>>(lnb, f1T, s1_f1b, nullptr, h1b, NTOK, 1024, 256);
  gemm_bf16<0,0><<<gP, blk, 0, stream>>>(h1b, f2T, s1_f2b, t2, t1, NTOK, 256, 1024); // y2

  // ---- swin block 2 (shift: roll+add, then MLP only) ----
  roll_add_k<<<(NTOK * 256) / 256, blk, 0, stream>>>(t1, t0);
  layernorm_k<<<NTOK / 8, blk, 0, stream>>>(t0, s2_n2g, s2_n2b, lnb);
  gemm_bf16<1,1><<<gF1, blk, 0, stream>>>(lnb, g1T, s2_f1b, nullptr, h1b, NTOK, 1024, 256);
  gemm_bf16<0,0><<<gP, blk, 0, stream>>>(h1b, g2T, s2_f2b, t0, (float*)d_out,
                                         NTOK, 256, 1024);
}
